// MultiHeadAttention_45363444580817
// MI455X (gfx1250) — matrix operand order repaired and checked by reading, not compiled, not run on hardware
//
#include <hip/hip_runtime.h>
#include <hip/hip_bf16.h>
#include <math.h>

// ---------------------------------------------------------------------------
// MI455X (gfx1250) fused multi-head attention forward.
// wave32, WMMA bf16 16x16x32 for all GEMMs, fp32 softmax/LN math.
// TDM (tensor_load_to_lds) stages activation panels into LDS.
// ---------------------------------------------------------------------------

typedef __attribute__((ext_vector_type(16))) __bf16   v16bf;
typedef __attribute__((ext_vector_type(8)))  float    v8f;
typedef __attribute__((ext_vector_type(4)))  uint32_t u32x4;
typedef __attribute__((ext_vector_type(8)))  uint32_t u32x8;

#define DMODEL 1024
#define NHEAD  16
#define DK     64
#define BATCH  2
#define SEQ    2048
#define ROWS   (BATCH * SEQ)   // 4096 flattened (b,s) rows

static __device__ __forceinline__ v8f wmma_bf16(v16bf a, v16bf b, v8f c) {
  // D = A(16x32 bf16) * B(32x16 bf16) + C(16x16 f32)
  return __builtin_amdgcn_wmma_f32_16x16x32_bf16(
      /*neg_a=*/false, a, /*neg_b=*/false, b,
      /*c_mod=*/(short)0, c, /*reuse_a=*/false, /*reuse_b=*/false);
}

// A-matrix 16x32 bf16 layout: lane (m = lane&15, half = lane>>4) holds
//   elems 0..7  -> K = 8*half + i ;  elems 8..15 -> K = 16 + 8*half + i
static __device__ __forceinline__ v16bf load_a_bf16(const __bf16* base, int ld,
                                                    int row0, int k0, int lane) {
  int m = lane & 15, half = lane >> 4;
  const __bf16* r = base + (size_t)(row0 + m) * ld + k0;
  v16bf a;
#pragma unroll
  for (int i = 0; i < 8; ++i) {
    a[i]     = r[8 * half + i];
    a[8 + i] = r[16 + 8 * half + i];
  }
  return a;
}

// B-matrix 32x16 bf16 tile where memory holds B^T rows contiguously:
//   B[k][n] = base[(r0 + n) * ld + k0 + k]
// -> per lane two 16B loads: K = 8*half + i and K = 16 + 8*half + i. Used for K^T, V^T, W^T tiles.
static __device__ __forceinline__ v16bf load_bT(const __bf16* base, int ld,
                                                int r0, int k0, int lane) {
  int n = lane & 15, half = lane >> 4;
  typedef __attribute__((ext_vector_type(8))) __bf16 v8bf;
  const __bf16* p = base + (size_t)(r0 + n) * ld + k0;
  v8bf lo = *(const v8bf*)(p + 8 * half);
  v8bf hi = *(const v8bf*)(p + 16 + 8 * half);
  return __builtin_shufflevector(lo, hi, 0, 1, 2, 3, 4, 5, 6, 7, 8, 9, 10, 11, 12, 13, 14, 15);
}

static __device__ __forceinline__ float rowmax16(float v) {
#pragma unroll
  for (int off = 8; off >= 1; off >>= 1) v = fmaxf(v, __shfl_xor(v, off, 32));
  return v;
}
static __device__ __forceinline__ float rowsum16(float v) {
#pragma unroll
  for (int off = 8; off >= 1; off >>= 1) v += __shfl_xor(v, off, 32);
  return v;
}

// TDM: async-load a 16-row x 2048-byte 2D tile (row stride 2048B) into LDS.
// Builds D# group0 (128b) / group1 (256b) in SGPRs, issues tensor_load_to_lds
// via inline asm (portable across ROCm/amdgpu-toolchain builtin arities).
static __device__ __forceinline__ void tdm_load_panel_16x2048B(
    uint32_t lds_off, const void* gptr) {
  uint64_t ga = (uint64_t)(uintptr_t)gptr;
  u32x4 g0;
  g0[0] = 1u;                                        // count=1, user mode
  g0[1] = lds_off;                                   // lds_addr
  g0[2] = (uint32_t)ga;                              // global_addr[31:0]
  g0[3] = (uint32_t)((ga >> 32) & 0x1FFFFFFu)        // global_addr[56:32]
          | (2u << 30);                              // type=2 ("image")
  // dwords per row = 512 (data_size=4B), 16 rows, stride = 512 dwords
  const uint32_t td0 = 512u, td1 = 16u, t0 = 512u, t1 = 16u;
  u32x8 g1;
  g1[0] = (2u << 16);                                // data_size = 4B
  g1[1] = (td0 & 0xFFFFu) << 16;                     // tensor_dim0[15:0]
  g1[2] = (td0 >> 16) | ((td1 & 0xFFFFu) << 16);     // td0 hi | tensor_dim1 lo
  g1[3] = (td1 >> 16) | (t0 << 16);                  // td1 hi | tile_dim0
  g1[4] = t1;                                        // tile_dim1 (tile_dim2=0)
  g1[5] = 512u;                                      // tensor_dim0_stride lo
  g1[6] = 0u;
  g1[7] = 0u;
  asm volatile("tensor_load_to_lds %0, %1" :: "s"(g0), "s"(g1) : "memory");
  __builtin_amdgcn_s_wait_tensorcnt(0);
  asm volatile("" ::: "memory");
}

// ---------------------------------------------------------------------------
// Prep kernels: one-shot fp32->bf16 conversion + operand transposes so every
// WMMA B tile is a pair of 16B per-lane loads later.
// ---------------------------------------------------------------------------
__global__ void __launch_bounds__(256)
cvt_kernel(const float* __restrict__ q, const float* __restrict__ k,
           const float* __restrict__ v, __bf16* __restrict__ qb,
           __bf16* __restrict__ kb, __bf16* __restrict__ vb) {
  const int which = blockIdx.y;
  const float* X = (which == 0) ? q : (which == 1) ? k : v;
  __bf16* Y = (which == 0) ? qb : (which == 1) ? kb : vb;
  size_t i0 = ((size_t)blockIdx.x * 256 + threadIdx.x) * 4;
  float4 xv = *(const float4*)(X + i0);
  Y[i0 + 0] = (__bf16)xv.x;
  Y[i0 + 1] = (__bf16)xv.y;
  Y[i0 + 2] = (__bf16)xv.z;
  Y[i0 + 3] = (__bf16)xv.w;
}

// WT[which][h][n][k] = W[which][h][k][n], bf16.  grid=(NHEAD,3), block=256.
__global__ void __launch_bounds__(256)
wT_kernel(const float* __restrict__ Wq, const float* __restrict__ Wk,
          const float* __restrict__ Wv, __bf16* __restrict__ WTall) {
  const int h = blockIdx.x, which = blockIdx.y;
  const float* W = ((which == 0) ? Wq : (which == 1) ? Wk : Wv)
                   + (size_t)h * DMODEL * DK;
  __bf16* O = WTall + ((size_t)which * NHEAD + h) * DK * DMODEL;
  for (int idx = threadIdx.x; idx < DMODEL * DK; idx += 256) {
    int kx = idx >> 6, nx = idx & 63;         // coalesced read over n
    O[(size_t)nx * DMODEL + kx] = (__bf16)W[idx];
  }
}

// WpT[n][k] = Wp[k][n], bf16.  grid=DMODEL (one k-row per block).
__global__ void __launch_bounds__(256)
wpT_kernel(const float* __restrict__ Wp, __bf16* __restrict__ WpT) {
  const int kx = blockIdx.x;
  for (int nx = threadIdx.x; nx < DMODEL; nx += 256)
    WpT[(size_t)nx * DMODEL + kx] = (__bf16)Wp[(size_t)kx * DMODEL + nx];
}

// ---------------------------------------------------------------------------
// Kernel 1: QKV projection. One block = one 16-row activation panel (TDM-
// staged in LDS, read once from HBM) x one tensor; 8 waves x 2 heads each.
// Outputs: qh,kh row-major [bh][s][64]; V stored transposed vhT [bh][64][s].
// grid = (ROWS/16, 3), block = 256.
// ---------------------------------------------------------------------------
__global__ void __launch_bounds__(256)
qkv_proj_kernel(const __bf16* __restrict__ qb, const __bf16* __restrict__ kb,
                const __bf16* __restrict__ vb, const __bf16* __restrict__ WTall,
                __bf16* __restrict__ qh, __bf16* __restrict__ kh,
                __bf16* __restrict__ vhT) {
  __shared__ __bf16 Apanel[16 * DMODEL];   // 32KB of the WGP's 320KB LDS

  const int row0  = blockIdx.x * 16;
  const int which = blockIdx.y;
  const __bf16* X  = (which == 0) ? qb : (which == 1) ? kb : vb;
  const __bf16* WT = WTall + (size_t)which * NHEAD * DK * DMODEL;

  if (threadIdx.x < 32) {   // wave 0 issues the TDM DMA for the whole block
    tdm_load_panel_16x2048B((uint32_t)(uintptr_t)&Apanel[0],
                            X + (size_t)row0 * DMODEL);
  }
  __syncthreads();

  const int lane = threadIdx.x & 31;
  const int wv   = threadIdx.x >> 5;     // 0..7
  const int n = lane & 15, half = lane >> 4;
  const int b = row0 / SEQ, s0 = row0 % SEQ;

  for (int hh = 0; hh < 2; ++hh) {
    const int h = wv * 2 + hh;
    const __bf16* Wh = WT + (size_t)h * DK * DMODEL;   // [64][1024] (W^T)
    v8f acc[4] = {{}, {}, {}, {}};
    for (int kk = 0; kk < DMODEL; kk += 32) {
      // A tile from LDS panel
      v16bf a;
      {
        const __bf16* r = &Apanel[(lane & 15) * DMODEL + kk];
#pragma unroll
        for (int i = 0; i < 8; ++i) {
          a[i]     = r[8 * half + i];
          a[8 + i] = r[16 + 8 * half + i];
        }
      }
#pragma unroll
      for (int c = 0; c < 4; ++c)
        acc[c] = wmma_bf16(a, load_bT(Wh, DMODEL, c * 16, kk, lane), acc[c]);
    }

    const size_t bh = (size_t)(b * NHEAD + h);
    if (which == 2) {        // V stored transposed: vhT[bh][col][s]
#pragma unroll
      for (int c = 0; c < 4; ++c)
#pragma unroll
        for (int r = 0; r < 8; ++r) {
          int M = r + 8 * half;
          vhT[(bh * DK + c * 16 + n) * SEQ + s0 + M] = (__bf16)acc[c][r];
        }
    } else {
      __bf16* Y = (which == 0) ? qh : kh;
#pragma unroll
      for (int c = 0; c < 4; ++c)
#pragma unroll
        for (int r = 0; r < 8; ++r) {
          int M = r + 8 * half;
          Y[(bh * SEQ + s0 + M) * DK + c * 16 + n] = (__bf16)acc[c][r];
        }
    }
  }
}

// ---------------------------------------------------------------------------
// Kernel 2: two-pass flash-style attention per (b,h, 16-row q tile).
// Pass 1: online rowmax/rowsum over all 2048 keys (scores discarded;
//         recompute beats re-reading 512MB of scores from HBM).
// Pass 2: recompute scores, write normalized probs to attns (head-major),
//         stage P tile as bf16 in LDS, WMMA-accumulate P @ V^T-tiles.
// grid = (SEQ/16, BATCH*NHEAD), block = 32 (one wave).
// ---------------------------------------------------------------------------
__global__ void __launch_bounds__(32)
attn_kernel(const __bf16* __restrict__ qh, const __bf16* __restrict__ kh,
            const __bf16* __restrict__ vhT, float* __restrict__ attns,
            __bf16* __restrict__ Ob) {
  __shared__ __bf16 Plds[16 * 32];

  const int lane = threadIdx.x;
  const int n = lane & 15, half = lane >> 4;
  const int q0 = blockIdx.x * 16;
  const int bh = blockIdx.y;              // b*NHEAD + h
  const int b = bh >> 4, h = bh & 15;

  const __bf16* qh_h = qh  + (size_t)bh * SEQ * DK;
  const __bf16* kh_h = kh  + (size_t)bh * SEQ * DK;
  const __bf16* vt_h = vhT + (size_t)bh * DK * SEQ;

  const v16bf a0 = load_a_bf16(qh_h, DK, q0, 0, lane);   // K=0..31
  const v16bf a1 = load_a_bf16(qh_h, DK, q0, 32, lane);  // K=32..63
  const float scale = 0.03125f;  // 1/sqrt(D_MODEL)

  float m[8], l[8];
#pragma unroll
  for (int r = 0; r < 8; ++r) { m[r] = -1e30f; l[r] = 0.0f; }

  // ---- pass 1: softmax statistics (online) ----
  for (int kt = 0; kt < SEQ; kt += 32) {
    if (kt + 128 < SEQ)
      __builtin_prefetch(kh_h + (size_t)(kt + 128) * DK, 0, 3);
    v8f c0 = {}, c1 = {};
    c0 = wmma_bf16(a0, load_bT(kh_h, DK, kt,      0,  lane), c0);
    c0 = wmma_bf16(a1, load_bT(kh_h, DK, kt,      32, lane), c0);
    c1 = wmma_bf16(a0, load_bT(kh_h, DK, kt + 16, 0,  lane), c1);
    c1 = wmma_bf16(a1, load_bT(kh_h, DK, kt + 16, 32, lane), c1);
#pragma unroll
    for (int r = 0; r < 8; ++r) {
      float s0v = c0[r] * scale, s1v = c1[r] * scale;
      float nm = fmaxf(m[r], rowmax16(fmaxf(s0v, s1v)));
      float rs = rowsum16(__expf(s0v - nm) + __expf(s1v - nm));
      l[r] = l[r] * __expf(m[r] - nm) + rs;
      m[r] = nm;
    }
  }
  float invl[8];
#pragma unroll
  for (int r = 0; r < 8; ++r) invl[r] = 1.0f / l[r];

  // ---- pass 2: recompute, emit probs, accumulate P @ V ----
  v8f oacc[4] = {{}, {}, {}, {}};
  const size_t arow = ((size_t)(h * BATCH + b) * SEQ + q0);  // attns row base

  for (int kt = 0; kt < SEQ; kt += 32) {
    v8f c0 = {}, c1 = {};
    c0 = wmma_bf16(a0, load_bT(kh_h, DK, kt,      0,  lane), c0);
    c0 = wmma_bf16(a1, load_bT(kh_h, DK, kt,      32, lane), c0);
    c1 = wmma_bf16(a0, load_bT(kh_h, DK, kt + 16, 0,  lane), c1);
    c1 = wmma_bf16(a1, load_bT(kh_h, DK, kt + 16, 32, lane), c1);

#pragma unroll
    for (int r = 0; r < 8; ++r) {
      int M = r + 8 * half;
      float p0 = __expf(c0[r] * scale - m[r]) * invl[r];
      float p1 = __expf(c1[r] * scale - m[r]) * invl[r];
      size_t row = (arow + M) * (size_t)SEQ + kt;
      attns[row + n]      = p0;   // mandatory 512MB output stream
      attns[row + 16 + n] = p1;
      Plds[M * 32 + n]      = (__bf16)p0;
      Plds[M * 32 + 16 + n] = (__bf16)p1;
    }
    __syncthreads();

    // P(16x32) in A-layout from LDS
    v16bf aP;
    {
      const __bf16* r = &Plds[n * 32];
#pragma unroll
      for (int i = 0; i < 8; ++i) {
        aP[i]     = r[8 * half + i];
        aP[8 + i] = r[16 + 8 * half + i];
      }
    }
#pragma unroll
    for (int c = 0; c < 4; ++c)
      oacc[c] = wmma_bf16(aP, load_bT(vt_h, SEQ, c * 16, kt, lane), oacc[c]);
    __syncthreads();
  }

  // store concat-head bf16 (consumed only as a WMMA A operand downstream)
#pragma unroll
  for (int c = 0; c < 4; ++c)
#pragma unroll
    for (int r = 0; r < 8; ++r) {
      int M = r + 8 * half;
      size_t row = (size_t)(b * SEQ + q0 + M);
      Ob[row * DMODEL + h * DK + c * 16 + n] = (__bf16)oacc[c][r];
    }
}

// ---------------------------------------------------------------------------
// Kernel 3: proj = Ob @ W_proj + b_proj + residual(q).  grid=(ROWS/16,16).
// ---------------------------------------------------------------------------
__global__ void __launch_bounds__(32)
out_proj_kernel(const __bf16* __restrict__ Ob, const __bf16* __restrict__ WpT,
                const float* __restrict__ bp, const float* __restrict__ qres,
                float* __restrict__ proj) {
  const int lane = threadIdx.x;
  const int row0 = blockIdx.x * 16;
  const int col0 = blockIdx.y * 64;

  v8f acc[4] = {{}, {}, {}, {}};
  for (int kk = 0; kk < DMODEL; kk += 32) {
    v16bf a = load_a_bf16(Ob, DMODEL, row0, kk, lane);
#pragma unroll
    for (int c = 0; c < 4; ++c)
      acc[c] = wmma_bf16(a, load_bT(WpT, DMODEL, col0 + c * 16, kk, lane),
                         acc[c]);
  }

  const int n = lane & 15, half = lane >> 4;
#pragma unroll
  for (int c = 0; c < 4; ++c)
#pragma unroll
    for (int r = 0; r < 8; ++r) {
      int M = r + 8 * half;
      int col = col0 + c * 16 + n;
      size_t idx = (size_t)(row0 + M) * DMODEL + col;
      proj[idx] = acc[c][r] + bp[col] + qres[idx];
    }
}

// ---------------------------------------------------------------------------
// Kernel 4: LayerNorm per row (torch semantics: ddof=1, eps added to std).
// ---------------------------------------------------------------------------
__global__ void __launch_bounds__(256)
ln_kernel(const float* __restrict__ x, const float* __restrict__ g,
          const float* __restrict__ be, float* __restrict__ y) {
  __shared__ float ss[8], ss2[8];
  __shared__ float mu_s, inv_s;

  const int row = blockIdx.x;
  const float* xr = x + (size_t)row * DMODEL;

  float s = 0.0f, s2 = 0.0f;
  for (int i = threadIdx.x; i < DMODEL; i += 256) {
    float v = xr[i];
    s += v; s2 += v * v;
  }
#pragma unroll
  for (int off = 16; off >= 1; off >>= 1) {
    s  += __shfl_xor(s, off, 32);
    s2 += __shfl_xor(s2, off, 32);
  }
  if ((threadIdx.x & 31) == 0) {
    ss[threadIdx.x >> 5] = s; ss2[threadIdx.x >> 5] = s2;
  }
  __syncthreads();
  if (threadIdx.x == 0) {
    float S = 0.0f, S2 = 0.0f;
#pragma unroll
    for (int i = 0; i < 8; ++i) { S += ss[i]; S2 += ss2[i]; }
    float mu = S / (float)DMODEL;
    float var = (S2 - (float)DMODEL * mu * mu) / (float)(DMODEL - 1);
    mu_s = mu;
    inv_s = 1.0f / (sqrtf(fmaxf(var, 0.0f)) + 1e-3f);
  }
  __syncthreads();
  const float mu = mu_s, inv = inv_s;
  for (int i = threadIdx.x; i < DMODEL; i += 256)
    y[(size_t)row * DMODEL + i] = (xr[i] - mu) * inv * g[i] + be[i];
}

// ---------------------------------------------------------------------------
extern "C" void kernel_launch(void* const* d_in, const int* in_sizes, int n_in,
                              void* d_out, int out_size, void* d_ws,
                              size_t ws_size, hipStream_t stream) {
  const float* q    = (const float*)d_in[0];
  const float* k    = (const float*)d_in[1];
  const float* v    = (const float*)d_in[2];
  const float* Wq   = (const float*)d_in[3];
  const float* Wk   = (const float*)d_in[4];
  const float* Wv   = (const float*)d_in[5];
  const float* Wp   = (const float*)d_in[6];
  const float* bp   = (const float*)d_in[7];
  const float* ln_a = (const float*)d_in[8];
  const float* ln_b = (const float*)d_in[9];

  float* y     = (float*)d_out;
  float* attns = y + (size_t)ROWS * DMODEL;  // [H*B, Sq, Sk] head-major

  // workspace layout (bf16 unless noted)
  char* ws = (char*)d_ws;
  const size_t actElems  = (size_t)ROWS * DMODEL;          // 4M
  const size_t headElems = (size_t)BATCH * NHEAD * SEQ * DK;
  __bf16* qb    = (__bf16*)ws;  ws += actElems * 2;        // 8MB
  __bf16* kb    = (__bf16*)ws;  ws += actElems * 2;
  __bf16* vb    = (__bf16*)ws;  ws += actElems * 2;
  __bf16* WTall = (__bf16*)ws;  ws += (size_t)3 * NHEAD * DK * DMODEL * 2; // 6MB
  __bf16* WpT   = (__bf16*)ws;  ws += (size_t)DMODEL * DMODEL * 2;         // 2MB
  __bf16* qh    = (__bf16*)ws;  ws += headElems * 2;       // 8MB
  __bf16* kh    = (__bf16*)ws;  ws += headElems * 2;
  __bf16* vhT   = (__bf16*)ws;  ws += headElems * 2;
  __bf16* Ob    = (__bf16*)ws;  ws += actElems * 2;        // 8MB
  float*  proj  = (float*)ws;                              // 16MB fp32

  cvt_kernel<<<dim3(actElems / (256 * 4), 3), 256, 0, stream>>>(
      q, k, v, qb, kb, vb);
  wT_kernel<<<dim3(NHEAD, 3), 256, 0, stream>>>(Wq, Wk, Wv, WTall);
  wpT_kernel<<<dim3(DMODEL), 256, 0, stream>>>(Wp, WpT);
  qkv_proj_kernel<<<dim3(ROWS / 16, 3), 256, 0, stream>>>(
      qb, kb, vb, WTall, qh, kh, vhT);
  attn_kernel<<<dim3(SEQ / 16, BATCH * NHEAD), 32, 0, stream>>>(
      qh, kh, vhT, attns, Ob);
  out_proj_kernel<<<dim3(ROWS / 16, DMODEL / 64), 32, 0, stream>>>(
      Ob, WpT, bp, q, proj);
  ln_kernel<<<dim3(ROWS), 256, 0, stream>>>(proj, ln_a, ln_b, y);
}
